// SelfAttention_6416681140452
// MI455X (gfx1250) — compile-verified, matrix operand order checked by reading, not run on hardware
//
#include <hip/hip_runtime.h>

// ---------------------------------------------------------------------------
// SAGAN self-attention, fused flash-attention style, bf16 WMMA on gfx1250,
// with double-buffered async global->LDS staging of K/V tiles (ASYNCcnt).
//   B=4, C=256, CQ=32, N=64*64=4096
// ---------------------------------------------------------------------------

#define B_DIM  4
#define C_DIM  256
#define CQ_DIM 32
#define N_DIM  4096

#define VPAD   72   // V LDS row stride (elems): 144B = 36-bank stride, no conflicts
#define KPAD   40   // K LDS row stride (elems): 80B  = 20-bank stride, no conflicts

typedef __attribute__((ext_vector_type(16))) __bf16 v16bf;
typedef __attribute__((ext_vector_type(8)))  __bf16 v8bf;
typedef __attribute__((ext_vector_type(8)))  float  v8f;

static __device__ inline v8f v8f_zero() {
    v8f z;
#pragma unroll
    for (int i = 0; i < 8; ++i) z[i] = 0.0f;
    return z;
}

// Assemble a 16-element bf16 A/B operand from two 16-byte loads.
// Per the CDNA5 16-bit A-matrix layout (16x32): lane owns K = kb..kb+7 in
// VGPR0-3 and K = 16+kb..23+kb in VGPR4-7, kb = (lane<16 ? 0 : 8).
static __device__ inline v16bf pack16(v8bf lo, v8bf hi) {
    union { struct { v8bf a, b; } s; v16bf v; } u;
    u.s.a = lo; u.s.b = hi;
    return u.v;
}

// Workgroup-relative LDS byte offset: generic LDS pointers on AMDGPU are
// aperture_hi32 ## as3_offset, so truncation yields the DS address.
static __device__ inline unsigned ldsoff(const void* p) {
    return (unsigned)(unsigned long long)p;
}

// Async global->LDS copy, 16B per lane; tracked by ASYNCcnt.
static __device__ inline void async_cp16(unsigned lds_byte_off, const void* gaddr) {
    asm volatile("global_load_async_to_lds_b128 %0, %1, off"
                 :: "v"(lds_byte_off), "v"(gaddr) : "memory");
}

// ---------------------------------------------------------------------------
// 16-lane all-reduce via single-instruction DPP16 VOP2 butterflies.
// ---------------------------------------------------------------------------
static __device__ inline float dpp_red_max16(float x) {
    float r;
    asm("v_max_num_f32_dpp %0, %1, %1 quad_perm:[1,0,3,2] row_mask:0xf bank_mask:0xf"
        : "=v"(r) : "v"(x));
    asm("v_max_num_f32_dpp %0, %1, %1 quad_perm:[2,3,0,1] row_mask:0xf bank_mask:0xf"
        : "=v"(x) : "v"(r));
    asm("v_max_num_f32_dpp %0, %1, %1 row_half_mirror row_mask:0xf bank_mask:0xf"
        : "=v"(r) : "v"(x));
    asm("v_max_num_f32_dpp %0, %1, %1 row_mirror row_mask:0xf bank_mask:0xf"
        : "=v"(x) : "v"(r));
    return x;
}

static __device__ inline float dpp_red_sum16(float x) {
    float r;
    asm("v_add_f32_dpp %0, %1, %1 quad_perm:[1,0,3,2] row_mask:0xf bank_mask:0xf"
        : "=v"(r) : "v"(x));
    asm("v_add_f32_dpp %0, %1, %1 quad_perm:[2,3,0,1] row_mask:0xf bank_mask:0xf"
        : "=v"(x) : "v"(r));
    asm("v_add_f32_dpp %0, %1, %1 row_half_mirror row_mask:0xf bank_mask:0xf"
        : "=v"(r) : "v"(x));
    asm("v_add_f32_dpp %0, %1, %1 row_mirror row_mask:0xf bank_mask:0xf"
        : "=v"(x) : "v"(r));
    return x;
}

// ---------------------------------------------------------------------------
// Kernel 1: q/k/v projections (+bias), fp32 GEMV tiles -> bf16 workspace.
//   Qb, Kb : [B, N, CQ]  (row-major over d)
//   Vb     : [B, C, N]   (channel-major)
// ---------------------------------------------------------------------------
__global__ __launch_bounds__(256) void proj_kernel(
    const float* __restrict__ x,
    const float* __restrict__ wq, const float* __restrict__ bq,
    const float* __restrict__ wk, const float* __restrict__ bk,
    const float* __restrict__ wv, const float* __restrict__ bv,
    unsigned short* __restrict__ Qb_,
    unsigned short* __restrict__ Kb_,
    unsigned short* __restrict__ Vb_)
{
    __shared__ float ws[8][C_DIM];
    __shared__ float bs[8];

    const int b  = blockIdx.z;
    const int g  = blockIdx.y;                 // row group: rows g*8 .. g*8+7
    const int n0 = blockIdx.x * 1024 + threadIdx.x * 4;
    const int r0 = g * 8;

    const float* w; const float* bias; int sel, rbase;
    if (r0 < CQ_DIM)          { w = wq; bias = bq; sel = 0; rbase = r0; }
    else if (r0 < 2 * CQ_DIM) { w = wk; bias = bk; sel = 1; rbase = r0 - CQ_DIM; }
    else                      { w = wv; bias = bv; sel = 2; rbase = r0 - 2 * CQ_DIM; }

    for (int i = threadIdx.x; i < 8 * C_DIM; i += 256)
        ws[i >> 8][i & 255] = w[(rbase + (i >> 8)) * C_DIM + (i & 255)];
    if (threadIdx.x < 8) bs[threadIdx.x] = bias[rbase + threadIdx.x];
    __syncthreads();

    float acc[8][4];
#pragma unroll
    for (int i = 0; i < 8; ++i)
#pragma unroll
        for (int j = 0; j < 4; ++j) acc[i][j] = 0.0f;

    const float* xb = x + (size_t)b * C_DIM * N_DIM + n0;
    for (int c = 0; c < C_DIM; ++c) {
        const float4 xv = *(const float4*)(xb + (size_t)c * N_DIM);
#pragma unroll
        for (int i = 0; i < 8; ++i) {
            const float wv_ = ws[i][c];
            acc[i][0] += wv_ * xv.x;
            acc[i][1] += wv_ * xv.y;
            acc[i][2] += wv_ * xv.z;
            acc[i][3] += wv_ * xv.w;
        }
    }

    if (sel == 2) {
        __bf16* Vb = reinterpret_cast<__bf16*>(Vb_) + (size_t)b * C_DIM * N_DIM;
#pragma unroll
        for (int i = 0; i < 8; ++i) {
            const float bv_ = bs[i];
            __bf16* vrow = Vb + (size_t)(rbase + i) * N_DIM + n0;
#pragma unroll
            for (int j = 0; j < 4; ++j)
                vrow[j] = (__bf16)(acc[i][j] + bv_);
        }
    } else {
        __bf16* T = reinterpret_cast<__bf16*>(sel ? Kb_ : Qb_) +
                    (size_t)b * N_DIM * CQ_DIM;
#pragma unroll
        for (int i = 0; i < 8; ++i) {
            const float bq_ = bs[i];
#pragma unroll
            for (int j = 0; j < 4; ++j)
                T[(size_t)(n0 + j) * CQ_DIM + rbase + i] = (__bf16)(acc[i][j] + bq_);
        }
    }
}

// ---------------------------------------------------------------------------
// Kernel 2: fused flash attention + residual, async-staged K/V tiles.
// Block: 256 threads = 8 waves; block owns 64 query rows of one batch.
// Wave (rg, half): 16 query rows x 128 output channels.
// Per key tile of 64: async-stage next K/V tile, 4 QK^T WMMAs from LDS,
// DPP online softmax, 16 P*V^T WMMAs from LDS.
// ---------------------------------------------------------------------------
__global__ __launch_bounds__(256) void flash_kernel(
    const unsigned short* __restrict__ Qb_,
    const unsigned short* __restrict__ Kb_,
    const unsigned short* __restrict__ Vb_,
    const float* __restrict__ x,
    const float* __restrict__ gamma,
    float* __restrict__ out)
{
    __shared__ __bf16 Vlds[2][C_DIM][VPAD];   // 2 x 36 KB staged V tile
    __shared__ __bf16 Klds[2][64][KPAD];      // 2 x  5 KB staged K tile
    __shared__ __bf16 Plds[8][16][64];        // per-wave P tile (16 KB)

    const int b    = blockIdx.y;
    const int q0   = blockIdx.x * 64;
    const int tid  = threadIdx.x;
    const int wave = tid >> 5;
    const int lane = tid & 31;
    const int rg   = wave >> 1;          // 0..3: query row group
    const int half = wave & 1;           // 0..1: channel half
    const int lm   = lane & 15;
    const int kb   = (lane >> 4) * 8;    // per-lane K-offset select
    const int ch0  = half * 128;

    const __bf16* Qb = reinterpret_cast<const __bf16*>(Qb_);
    const __bf16* Kb = reinterpret_cast<const __bf16*>(Kb_);
    const __bf16* Vb = reinterpret_cast<const __bf16*>(Vb_);

    // Q A-operand: 16 rows x 32 (full head dim), resident for whole kernel.
    const __bf16* qrow = Qb + ((size_t)b * N_DIM + q0 + rg * 16 + lm) * CQ_DIM;
    const v16bf qa = pack16(*(const v8bf*)(qrow + kb),
                            *(const v8bf*)(qrow + 16 + kb));

    const __bf16* Kbase = Kb + (size_t)b * N_DIM * CQ_DIM;
    const __bf16* Vbase = Vb + (size_t)b * C_DIM * N_DIM;

    // Cooperative async staging of one 64-key tile (V: 256x64, K: 64x32).
    const int vr = tid >> 3;            // V row group base (0..31)
    const int vc = (tid & 7) * 8;       // V element chunk (0..56)
    const int kr = tid >> 2;            // K row (0..63)
    const int kc = (tid & 3) * 8;       // K element chunk (0..24)

#define STAGE_TILE(BUF, J0)                                                  \
    do {                                                                     \
        _Pragma("unroll")                                                    \
        for (int p = 0; p < 8; ++p) {                                        \
            const int row = vr + p * 32;                                     \
            async_cp16(ldsoff(&Vlds[BUF][row][vc]),                          \
                       Vbase + (size_t)row * N_DIM + (J0) + vc);             \
        }                                                                    \
        async_cp16(ldsoff(&Klds[BUF][kr][kc]),                               \
                   Kbase + (size_t)((J0) + kr) * CQ_DIM + kc);               \
    } while (0)

    float m_run[8], l_run[8];
    v8f   o_acc[8];
#pragma unroll
    for (int v = 0; v < 8; ++v) {
        m_run[v] = -3.0e38f;
        l_run[v] = 0.0f;
        o_acc[v] = v8f_zero();
    }

    // Prologue: stage tile 0, wait, sync.
    STAGE_TILE(0, 0);
    asm volatile("s_wait_asynccnt 0" ::: "memory");
    __syncthreads();

    for (int jt = 0; jt < N_DIM / 64; ++jt) {
        const int j0  = jt * 64;
        const int cur = jt & 1;
        const int nxt = cur ^ 1;
        const bool more = (j0 + 64) < N_DIM;

        // ---- kick off async staging of the next tile into the other buffer
        if (more) STAGE_TILE(nxt, j0 + 64);

        // ---- S = Q K^T over 4 column sub-tiles of 16 keys each (K from LDS)
        v8f S[4];
#pragma unroll
        for (int t = 0; t < 4; ++t) {
            const __bf16* kp = &Klds[cur][t * 16 + lm][0];
            const v16bf kmat = pack16(*(const v8bf*)(kp + kb),
                                      *(const v8bf*)(kp + 16 + kb));
            S[t] = __builtin_amdgcn_wmma_f32_16x16x32_bf16(
                false, qa, false, kmat, (short)0, v8f_zero(), false, false);
        }

        // ---- online softmax (row = C-layout VGPR v, spans 16 lanes)
#pragma unroll
        for (int v = 0; v < 8; ++v) {
            float mx = fmaxf(fmaxf(S[0][v], S[1][v]), fmaxf(S[2][v], S[3][v]));
            mx = dpp_red_max16(mx);
            const float m_new = fmaxf(m_run[v], mx);
            const float alpha = __expf(m_run[v] - m_new);
            m_run[v] = m_new;

            float rs = 0.0f;
#pragma unroll
            for (int t = 0; t < 4; ++t) {
                const float p = __expf(S[t][v] - m_new);
                S[t][v] = p;
                rs += p;
            }
            rs = dpp_red_sum16(rs);
            l_run[v] = l_run[v] * alpha + rs;
#pragma unroll
            for (int ct = 0; ct < 8; ++ct)
                o_acc[ct][v] *= alpha;
        }

        // ---- P (C layout) -> LDS as row-major bf16 [16][64], per-wave region
        const int rofs = (lane >> 4) * 8;
#pragma unroll
        for (int v = 0; v < 8; ++v)
#pragma unroll
            for (int t = 0; t < 4; ++t)
                Plds[wave][v + rofs][t * 16 + lm] = (__bf16)S[t][v];

        // Same-wave DS ops are in-order; make the store->load edge explicit.
        asm volatile("s_wait_dscnt 0" ::: "memory");

        // ---- reload P as two A operands (K = 0..31 and K = 32..63)
        const __bf16* prow = &Plds[wave][lm][0];
        const v16bf pa0 = pack16(*(const v8bf*)(prow + kb),
                                 *(const v8bf*)(prow + 16 + kb));
        const v16bf pa1 = pack16(*(const v8bf*)(prow + 32 + kb),
                                 *(const v8bf*)(prow + 48 + kb));

        // ---- O += P V^T over this wave's 128 channels (V from LDS)
#pragma unroll
        for (int ct = 0; ct < 8; ++ct) {
            const __bf16* vp = &Vlds[cur][ch0 + ct * 16 + lm][0];
            const v16bf vb0 = pack16(*(const v8bf*)(vp + kb),
                                     *(const v8bf*)(vp + 16 + kb));
            o_acc[ct] = __builtin_amdgcn_wmma_f32_16x16x32_bf16(
                false, pa0, false, vb0, (short)0, o_acc[ct], false, false);
            const v16bf vb1 = pack16(*(const v8bf*)(vp + 32 + kb),
                                     *(const v8bf*)(vp + 48 + kb));
            o_acc[ct] = __builtin_amdgcn_wmma_f32_16x16x32_bf16(
                false, pa1, false, vb1, (short)0, o_acc[ct], false, false);
        }

        // ---- retire this iteration: my async copies done, then block-wide
        //      barrier so nobody overwrites a buffer still being read.
        if (more) {
            asm volatile("s_wait_asynccnt 0" ::: "memory");
            __syncthreads();
        }
    }
#undef STAGE_TILE

    // ---- epilogue: out = gamma * (O / l) + x
    const float g = gamma[0];
    const float* xb = x   + (size_t)b * C_DIM * N_DIM;
    float*       ob = out + (size_t)b * C_DIM * N_DIM;
#pragma unroll
    for (int v = 0; v < 8; ++v) {
        const int   n   = q0 + rg * 16 + v + ((lane >> 4) << 3);
        const float inv = 1.0f / l_run[v];
#pragma unroll
        for (int ct = 0; ct < 8; ++ct) {
            const int    o   = ch0 + ct * 16 + lm;
            const size_t idx = (size_t)o * N_DIM + n;
            ob[idx] = g * (o_acc[ct][v] * inv) + xb[idx];
        }
    }
}

// ---------------------------------------------------------------------------
extern "C" void kernel_launch(void* const* d_in, const int* in_sizes, int n_in,
                              void* d_out, int out_size, void* d_ws, size_t ws_size,
                              hipStream_t stream) {
    const float* x     = (const float*)d_in[0];
    const float* wq    = (const float*)d_in[1];
    const float* bq    = (const float*)d_in[2];
    const float* wk    = (const float*)d_in[3];
    const float* bk    = (const float*)d_in[4];
    const float* wv    = (const float*)d_in[5];
    const float* bv    = (const float*)d_in[6];
    const float* gamma = (const float*)d_in[7];
    float* out = (float*)d_out;

    // Workspace layout (bf16 stored as u16): Qb | Kb | Vb  = 1MB + 1MB + 8MB
    unsigned short* Qb = (unsigned short*)d_ws;
    unsigned short* Kb = Qb + (size_t)B_DIM * N_DIM * CQ_DIM;
    unsigned short* Vb = Kb + (size_t)B_DIM * N_DIM * CQ_DIM;

    dim3 gp(N_DIM / 1024, (2 * CQ_DIM + C_DIM) / 8, B_DIM);
    proj_kernel<<<gp, 256, 0, stream>>>(x, wq, bq, wk, bk, wv, bv, Qb, Kb, Vb);

    dim3 ga(N_DIM / 64, B_DIM);
    flash_kernel<<<ga, 256, 0, stream>>>(Qb, Kb, Vb, x, gamma, out);
}
